// TransformerModel_40819369181972
// MI455X (gfx1250) — matrix operand order repaired and checked by reading, not compiled, not run on hardware
//
#include <hip/hip_runtime.h>
#include <hip/hip_bf16.h>

// ---------------------------------------------------------------------------
// GPT-style transformer forward for gfx1250 (MI455X), all GEMMs via
// v_wmma_f32_16x16x32_f16 (wave32). GEMM: 4 M-tiles per wave, B-fragment reuse.
// ---------------------------------------------------------------------------

#define D_   1024
#define H_   8
#define L_   6
#define S_   1024
#define B_   2
#define HD_  128
#define FF_  4096
#define V_   50257
#define R_   (S_ * B_)   // 2048 rows, r = s*B+b

typedef __attribute__((ext_vector_type(16))) _Float16 v16h;
typedef __attribute__((ext_vector_type(8)))  _Float16 v8h;
typedef __attribute__((ext_vector_type(8)))  float    v8f;

__device__ __forceinline__ v16h make_v16(v8h lo, v8h hi) {
    v16h r;
#pragma unroll
    for (int i = 0; i < 8; ++i) { r[i] = lo[i]; r[i + 8] = hi[i]; }
    return r;
}

// A fragment: 16x32 f16, row-major source, lda in elements.
// lane<16: M=lane, K = k0..k0+15
// lane>=16: M=lane-16, K = k0+16..k0+31
__device__ __forceinline__ v16h load_A(const _Float16* __restrict__ A, int lda,
                                       int m0, int k0, int lane) {
    int half = lane >> 4, m = m0 + (lane & 15);
    const _Float16* p = A + (size_t)m * lda + k0 + half * 16;
    v8h lo = *(const v8h*)p;
    v8h hi = *(const v8h*)(p + 8);
    return make_v16(lo, hi);
}

// B fragment: 32x16 f16 from transposed weights Bt[N][K].
// lane<16: N=n0+lane, K = k0..k0+15 ; lane>=16: N=n0+lane-16, K = k0+16..k0+31
__device__ __forceinline__ v16h load_Bt(const _Float16* __restrict__ Bt, int ldb,
                                        int n0, int k0, int lane) {
    int half = lane >> 4, n = n0 + (lane & 15);
    const _Float16* p = Bt + (size_t)n * ldb + k0 + half * 16;
    v8h lo = *(const v8h*)p;
    v8h hi = *(const v8h*)(p + 8);
    return make_v16(lo, hi);
}

__device__ __forceinline__ v16h load_Bt_guard(const _Float16* __restrict__ Bt, int ldb,
                                              int n0, int k0, int lane, int N) {
    int half = lane >> 4, n = n0 + (lane & 15);
    if (n >= N) { v16h z = {}; return z; }
    const _Float16* p = Bt + (size_t)n * ldb + k0 + half * 16;
    v8h lo = *(const v8h*)p;
    v8h hi = *(const v8h*)(p + 8);
    return make_v16(lo, hi);
}

// ---------------------------------------------------------------------------
// Elementwise / prep kernels
// ---------------------------------------------------------------------------

__global__ __launch_bounds__(256) void embed_kernel(const int* __restrict__ tokens,
                                                    const float* __restrict__ tokE,
                                                    const float* __restrict__ posE,
                                                    float* __restrict__ x) {
    int r = blockIdx.x;            // r = s*B + b
    int s = r >> 1;                // B_ == 2
    int t = tokens[r];
    const float* te = tokE + (size_t)t * D_;
    const float* pe = posE + (size_t)s * D_;
    float* xr = x + (size_t)r * D_;
    for (int d = threadIdx.x; d < D_; d += 256) xr[d] = te[d] + pe[d];
}

__global__ __launch_bounds__(256) void rmsnorm_f16_kernel(const float* __restrict__ x,
                                                          const float* __restrict__ w,
                                                          _Float16* __restrict__ out) {
    __shared__ float red[256];
    int r = blockIdx.x;
    const float* xr = x + (size_t)r * D_;
    float ss = 0.f;
    for (int d = threadIdx.x; d < D_; d += 256) { float v = xr[d]; ss += v * v; }
    red[threadIdx.x] = ss;
    __syncthreads();
    for (int s = 128; s > 0; s >>= 1) {
        if (threadIdx.x < s) red[threadIdx.x] += red[threadIdx.x + s];
        __syncthreads();
    }
    float rinv = rsqrtf(red[0] * (1.0f / D_) + 1e-5f);
    _Float16* o = out + (size_t)r * D_;
    for (int d = threadIdx.x; d < D_; d += 256)
        o[d] = (_Float16)(w[d] * xr[d] * rinv);
}

// W (H, D, HD) fp32 -> Bt f16 [H*HD][D]
__global__ __launch_bounds__(256) void wt_qkv_kernel(const float* __restrict__ W,
                                                     _Float16* __restrict__ out) {
    int j = blockIdx.x;                 // 0..1023 output row (h*HD + c)
    int h = j >> 7, c = j & 127;
    const float* src = W + (size_t)h * D_ * HD_ + c;
    _Float16* o = out + (size_t)j * D_;
    for (int d = threadIdx.x; d < D_; d += 256)
        o[d] = (_Float16)src[(size_t)d * HD_];
}

// W (K, N) fp32 row-major -> Bt f16 [N][K]
__global__ __launch_bounds__(256) void wt_plain_kernel(const float* __restrict__ W,
                                                       _Float16* __restrict__ out,
                                                       int K, int N) {
    int n = blockIdx.x;
    _Float16* o = out + (size_t)n * K;
    for (int k = threadIdx.x; k < K; k += 256)
        o[k] = (_Float16)W[(size_t)k * N + n];
}

__global__ __launch_bounds__(256) void f32tof16_kernel(const float* __restrict__ in,
                                                       _Float16* __restrict__ out,
                                                       long long n) {
    long long i = (long long)blockIdx.x * 256 + threadIdx.x;
    long long stride = (long long)gridDim.x * 256;
    for (; i < n; i += stride) out[i] = (_Float16)in[i];
}

// ---------------------------------------------------------------------------
// Generic WMMA GEMM: C[M,N] (+)= A[M,K](f16) * Bt[N,K](f16)^T
// Block: 256 threads = 8 waves. Each wave owns a 64(M) x 16(N) strip:
// one B fragment per K-chunk is reused across 4 stacked WMMAs (2.5 loads/WMMA).
// Grid: x = ceil(N/128), y = M/64.
// ---------------------------------------------------------------------------
enum { MODE_F16 = 0, MODE_F16_SILU = 1, MODE_F32_STORE = 2, MODE_F32_ADD = 3 };
#define MT_ 4   // M-tiles per wave

template <int MODE>
__global__ __launch_bounds__(256) void gemm_kernel(const _Float16* __restrict__ A,
                                                   const _Float16* __restrict__ Bt,
                                                   void* __restrict__ Out,
                                                   int M, int N, int K) {
    int lane = threadIdx.x & 31;
    int wave = threadIdx.x >> 5;
    int n0 = blockIdx.x * 128 + wave * 16;
    int m0 = blockIdx.y * (16 * MT_);
    if (n0 >= N) return;
    bool tail = (n0 + 16 > N);

    v8f acc[MT_];
#pragma unroll
    for (int t = 0; t < MT_; ++t) acc[t] = (v8f){};

    const _Float16* bpref = Bt + (size_t)(n0 + (lane & 15)) * K;
    for (int k0 = 0; k0 < K; k0 += 32) {
        // prefetch B stream one chunk ahead (global_prefetch_b8)
        if (k0 + 32 < K) __builtin_prefetch(bpref + k0 + 32, 0, 1);
        v16h b = tail ? load_Bt_guard(Bt, K, n0, k0, lane, N)
                      : load_Bt(Bt, K, n0, k0, lane);
#pragma unroll
        for (int t = 0; t < MT_; ++t) {
            v16h a = load_A(A, K, m0 + t * 16, k0, lane);
            acc[t] = __builtin_amdgcn_wmma_f32_16x16x32_f16(false, a, false, b,
                                                            (short)0, acc[t], false, false);
        }
    }

    int col = n0 + (lane & 15);
    int mh  = (lane >> 4) << 3;
#pragma unroll
    for (int t = 0; t < MT_; ++t) {
#pragma unroll
        for (int i = 0; i < 8; ++i) {
            int m = m0 + t * 16 + i + mh;
            if (col < N) {
                float v = acc[t][i];
                size_t idx = (size_t)m * N + col;
                if constexpr (MODE == MODE_F16) {
                    ((_Float16*)Out)[idx] = (_Float16)v;
                } else if constexpr (MODE == MODE_F16_SILU) {
                    float s = v / (1.0f + __expf(-v));
                    ((_Float16*)Out)[idx] = (_Float16)s;
                } else if constexpr (MODE == MODE_F32_STORE) {
                    ((float*)Out)[idx] = v;
                } else {
                    ((float*)Out)[idx] += v;
                }
            }
        }
    }
}

// ---------------------------------------------------------------------------
// Attention: one block per (b,h, 16-query tile). Scores in LDS, block softmax,
// P (f16, LDS) x V via WMMA.
// ---------------------------------------------------------------------------
#define SM_SCORES   0                              // 16*1024 f32 = 64 KB
#define SM_PBUF     (16 * 1024 * 4)                // 16*1024 f16 = 32 KB
#define SM_RED      (SM_PBUF + 16 * 1024 * 2)      // 256 f32
#define SM_RSTAT    (SM_RED + 256 * 4)             // 16 f32
#define SM_TOTAL    (SM_RSTAT + 16 * 4)

__global__ __launch_bounds__(256) void attn_kernel(const _Float16* __restrict__ q,
                                                   const _Float16* __restrict__ k,
                                                   const _Float16* __restrict__ v,
                                                   _Float16* __restrict__ attn) {
    extern __shared__ char smem[];
    float*    scores = (float*)(smem + SM_SCORES);
    _Float16* pbuf   = (_Float16*)(smem + SM_PBUF);
    float*    red    = (float*)(smem + SM_RED);
    float*    rstat  = (float*)(smem + SM_RSTAT);

    const int qt = blockIdx.x;         // query tile 0..63
    const int bh = blockIdx.y;         // 0..15
    const int b = bh >> 3, h = bh & 7;
    const int sbase = qt * 16;
    const int tmax  = sbase + 16;      // keys in [0, tmax)
    const int lane = threadIdx.x & 31, wave = threadIdx.x >> 5;
    const float scale = 0.08838834764831845f;   // 1/sqrt(128)

    // --- q A-fragments (16 x 128 split into 4 K-chunks of 32) ---
    const int mhalf = lane >> 4, mr = lane & 15;
    const int sA = sbase + mr;
    const _Float16* qrow = q + (size_t)(sA * B_ + b) * D_ + h * HD_;
    v16h aq[4];
#pragma unroll
    for (int c = 0; c < 4; ++c) {
        const _Float16* p = qrow + c * 32 + mhalf * 16;
        aq[c] = make_v16(*(const v8h*)p, *(const v8h*)(p + 8));
    }

    // --- scores = scale * q @ k^T, causal-masked, into LDS ---
    for (int t0 = wave * 16; t0 < tmax; t0 += 8 * 16) {
        int key = t0 + (lane & 15);
        const _Float16* krow = k + (size_t)(key * B_ + b) * D_ + h * HD_;
        v8f acc = {};
#pragma unroll
        for (int c = 0; c < 4; ++c) {
            const _Float16* p = krow + c * 32 + (lane >> 4) * 16;
            v16h bf = make_v16(*(const v8h*)p, *(const v8h*)(p + 8));
            acc = __builtin_amdgcn_wmma_f32_16x16x32_f16(false, aq[c], false, bf,
                                                         (short)0, acc, false, false);
        }
        int col = lane & 15;
        int mh = (lane >> 4) << 3;
#pragma unroll
        for (int i = 0; i < 8; ++i) {
            int m = i + mh;
            int t = t0 + col;
            float sv = acc[i] * scale;
            if (t > sbase + m) sv = -__builtin_inff();
            scores[m * 1024 + t] = sv;
        }
    }
    __syncthreads();

    // --- block softmax over [0, tmax) per row ---
    int row = threadIdx.x >> 4, sub = threadIdx.x & 15;
    float mx = -__builtin_inff();
    for (int t = sub; t < tmax; t += 16) mx = fmaxf(mx, scores[row * 1024 + t]);
    red[threadIdx.x] = mx;
    __syncthreads();
    mx = -__builtin_inff();
#pragma unroll
    for (int i = 0; i < 16; ++i) mx = fmaxf(mx, red[row * 16 + i]);
    __syncthreads();                   // everyone done reading red
    float sum = 0.f;
    for (int t = sub; t < tmax; t += 16) {
        float e = __expf(scores[row * 1024 + t] - mx);
        pbuf[row * 1024 + t] = (_Float16)e;
        sum += e;
    }
    red[threadIdx.x] = sum;
    __syncthreads();
    if (sub == 0) {
        float s2 = 0.f;
#pragma unroll
        for (int i = 0; i < 16; ++i) s2 += red[row * 16 + i];
        rstat[row] = 1.0f / s2;
    }
    __syncthreads();

    // --- out(16x128) = P(16 x tmax) @ V(tmax x 128); wave w -> hd tile w*16 ---
    int n0 = wave * 16;
    v8f acc = {};
    for (int c = 0; c * 32 < tmax; ++c) {
        // A fragment from LDS pbuf (groups of 8 are either fully valid or fully past tmax)
        int kb0 = c * 32 + mhalf * 16;
        v8h zer = {};
        const _Float16* pa = pbuf + mr * 1024 + kb0;
        v8h lo = (kb0 < tmax) ? *(const v8h*)pa : zer;
        v8h hi = (kb0 + 8 < tmax) ? *(const v8h*)(pa + 8) : zer;
        v16h ap = make_v16(lo, hi);
        // B fragment: column hd = n0+(lane&15), K = key index (strided gather from v)
        v16h bv;
        int ncol = n0 + (lane & 15);
        int kb = c * 32 + (lane >> 4) * 16;
#pragma unroll
        for (int j = 0; j < 16; ++j) {
            int t = kb + j;
            bv[j] = (t < tmax) ? v[(size_t)(t * B_ + b) * D_ + h * HD_ + ncol]
                               : (_Float16)0.0f;
        }
        acc = __builtin_amdgcn_wmma_f32_16x16x32_f16(false, ap, false, bv,
                                                     (short)0, acc, false, false);
    }
    int ncol = n0 + (lane & 15);
    int mh = (lane >> 4) << 3;
#pragma unroll
    for (int i = 0; i < 8; ++i) {
        int m = i + mh;
        int s = sbase + m;
        float o = acc[i] * rstat[m];
        attn[(size_t)(s * B_ + b) * D_ + h * HD_ + ncol] = (_Float16)o;
    }
}

// ---------------------------------------------------------------------------
// Host orchestration
// ---------------------------------------------------------------------------
extern "C" void kernel_launch(void* const* d_in, const int* in_sizes, int n_in,
                              void* d_out, int out_size, void* d_ws, size_t ws_size,
                              hipStream_t stream) {
    const int*   tokens      = (const int*)d_in[0];
    const float* tok_emb     = (const float*)d_in[1];
    const float* pos_emb     = (const float*)d_in[2];
    const float* attn_norm_w = (const float*)d_in[3];
    const float* Wq          = (const float*)d_in[4];
    const float* Wk          = (const float*)d_in[5];
    const float* Wv          = (const float*)d_in[6];
    const float* Wo          = (const float*)d_in[7];
    const float* mlp_norm_w  = (const float*)d_in[8];
    const float* W1          = (const float*)d_in[9];
    const float* W2          = (const float*)d_in[10];
    const float* final_norm_w= (const float*)d_in[11];
    float* logits = (float*)d_out;

    char* ws = (char*)d_ws;
    size_t off = 0;
    float*    x     = (float*)(ws + off);    off += (size_t)R_ * D_ * 4;   // 8 MB
    _Float16* n16   = (_Float16*)(ws + off); off += (size_t)R_ * D_ * 2;
    _Float16* q16   = (_Float16*)(ws + off); off += (size_t)R_ * D_ * 2;
    _Float16* k16   = (_Float16*)(ws + off); off += (size_t)R_ * D_ * 2;
    _Float16* v16   = (_Float16*)(ws + off); off += (size_t)R_ * D_ * 2;
    _Float16* att16 = (_Float16*)(ws + off); off += (size_t)R_ * D_ * 2;
    _Float16* h16   = (_Float16*)(ws + off); off += (size_t)R_ * FF_ * 2;  // 16 MB
    _Float16* WqT   = (_Float16*)(ws + off); off += (size_t)D_ * D_ * 2;
    _Float16* WkT   = (_Float16*)(ws + off); off += (size_t)D_ * D_ * 2;
    _Float16* WvT   = (_Float16*)(ws + off); off += (size_t)D_ * D_ * 2;
    _Float16* WoT   = (_Float16*)(ws + off); off += (size_t)D_ * D_ * 2;
    _Float16* W1T   = (_Float16*)(ws + off); off += (size_t)D_ * FF_ * 2;
    _Float16* W2T   = (_Float16*)(ws + off); off += (size_t)FF_ * D_ * 2;
    _Float16* tokEh = (_Float16*)(ws + off); off += (size_t)V_ * D_ * 2;   // ~103 MB

    dim3 blk(256);

    // embeddings + f16 vocab table
    embed_kernel<<<R_, blk, 0, stream>>>(tokens, tok_emb, pos_emb, x);
    f32tof16_kernel<<<4096, blk, 0, stream>>>(tok_emb, tokEh, (long long)V_ * D_);

    dim3 g_gemm_d(D_ / 128, R_ / 64);      // (8, 32)
    dim3 g_gemm_ff(FF_ / 128, R_ / 64);    // (32, 32)
    dim3 g_attn(S_ / 16, B_ * H_);         // (64, 16)

    for (int l = 0; l < L_; ++l) {
        const size_t wOff = (size_t)l * D_ * D_;        // H*D*HD == D*D
        const size_t ffOff = (size_t)l * D_ * FF_;

        // attention block
        rmsnorm_f16_kernel<<<R_, blk, 0, stream>>>(x, attn_norm_w + (size_t)l * D_, n16);
        wt_qkv_kernel<<<D_, blk, 0, stream>>>(Wq + wOff, WqT);
        wt_qkv_kernel<<<D_, blk, 0, stream>>>(Wk + wOff, WkT);
        wt_qkv_kernel<<<D_, blk, 0, stream>>>(Wv + wOff, WvT);
        wt_plain_kernel<<<D_, blk, 0, stream>>>(Wo + wOff, WoT, D_, D_);

        gemm_kernel<MODE_F16><<<g_gemm_d, blk, 0, stream>>>(n16, WqT, q16, R_, D_, D_);
        gemm_kernel<MODE_F16><<<g_gemm_d, blk, 0, stream>>>(n16, WkT, k16, R_, D_, D_);
        gemm_kernel<MODE_F16><<<g_gemm_d, blk, 0, stream>>>(n16, WvT, v16, R_, D_, D_);

        attn_kernel<<<g_attn, blk, SM_TOTAL, stream>>>(q16, k16, v16, att16);

        gemm_kernel<MODE_F32_ADD><<<g_gemm_d, blk, 0, stream>>>(att16, WoT, x, R_, D_, D_);

        // MLP block
        rmsnorm_f16_kernel<<<R_, blk, 0, stream>>>(x, mlp_norm_w + (size_t)l * D_, n16);
        wt_plain_kernel<<<FF_, blk, 0, stream>>>(W1 + ffOff, W1T, D_, FF_);
        wt_plain_kernel<<<D_, blk, 0, stream>>>(W2 + ffOff, W2T, FF_, D_);

        gemm_kernel<MODE_F16_SILU><<<g_gemm_ff, blk, 0, stream>>>(n16, W1T, h16, R_, FF_, D_);
        gemm_kernel<MODE_F32_ADD><<<g_gemm_d, blk, 0, stream>>>(h16, W2T, x, R_, D_, FF_);
    }

    // final norm + tied lm_head
    rmsnorm_f16_kernel<<<R_, blk, 0, stream>>>(x, final_norm_w, n16);
    dim3 g_logits((V_ + 127) / 128, R_ / 64);   // (393, 32)
    gemm_kernel<MODE_F32_STORE><<<g_logits, blk, 0, stream>>>(n16, tokEh, logits, R_, V_, D_);

    (void)in_sizes; (void)n_in; (void)out_size; (void)ws_size;
}
